// RPN_84937273245853
// MI455X (gfx1250) — compile-verified, matrix operand order checked by reading, not run on hardware
//
#include <hip/hip_runtime.h>
#include <hip/hip_bf16.h>
#include <math.h>
#include <stdint.h>

// ---------------------------------------------------------------------------
// Types for CDNA5 WMMA (gfx1250, wave32)
// ---------------------------------------------------------------------------
typedef __bf16 bf16_t;
typedef __attribute__((ext_vector_type(16))) __bf16 v16bf;
typedef __attribute__((ext_vector_type(8)))  __bf16 v8bf;
typedef __attribute__((ext_vector_type(8)))  float  v8f;
typedef __attribute__((ext_vector_type(4)))  unsigned int v4u;

union Frag16 { v16bf v; v8bf h[2]; };

// ---------------------------------------------------------------------------
// Problem constants
// ---------------------------------------------------------------------------
#define NB      8
#define CIN     512
#define COUT    512
#define FH      64
#define FW      64
#define NPIX    (FH*FW)          // 4096
#define PW      66               // padded width/height (1-pixel halo)
#define NA      9
#define NCAND   (NPIX*NA)        // 36864
#define N_PRE   50
#define N_POST  10
#define NMS_IOU 0.7f
#define MIN_SZ  16.0f

// d_out float offsets (return-order concatenation)
#define OFF_LOCS   0
#define OFF_SCORES 1179648      // 8*36864*4
#define OFF_ROIS   1769472
#define OFF_RIDX   1769792
#define OFF_ANCH   1769872

// CDNA5 async memory->LDS copy (ASYNCcnt path); lds = 32-bit LDS byte address
__device__ __forceinline__ void async_copy_b128(unsigned lds, const void* gaddr) {
    asm volatile("global_load_async_to_lds_b128 %0, %1, off"
                 :: "v"(lds), "v"(gaddr) : "memory");
}
__device__ __forceinline__ void wait_async0() {
    asm volatile("s_wait_asynccnt 0x0" ::: "memory");
}

// ---------------------------------------------------------------------------
// Prep: zero the padded NHWC buffer (halo stays zero)
// ---------------------------------------------------------------------------
__global__ void zero_xhp_kernel(v4u* __restrict__ p, int n16) {
    int i = blockIdx.x * 256 + threadIdx.x;
    if (i < n16) { v4u z = {0u, 0u, 0u, 0u}; p[i] = z; }
}

// Prep: x (NCHW f32) -> xhp (padded NHWC bf16), LDS-tiled transpose
__global__ __launch_bounds__(256) void prep_xh_kernel(const float* __restrict__ x,
                                                      bf16_t* __restrict__ xhp) {
    __shared__ float tile[32][33];
    const int img = blockIdx.z;
    const int p0 = blockIdx.x * 32, c0 = blockIdx.y * 32;
    const int tx = threadIdx.x & 31, ty = threadIdx.x >> 5;   // 32 x 8
    #pragma unroll
    for (int i = 0; i < 32; i += 8)
        tile[ty + i][tx] = x[((size_t)img * CIN + c0 + ty + i) * NPIX + p0 + tx];
    __syncthreads();
    #pragma unroll
    for (int i = 0; i < 32; i += 8) {
        const int p = p0 + ty + i;
        const int yy = p >> 6, xx = p & 63;
        xhp[(((size_t)img * PW + yy + 1) * PW + xx + 1) * CIN + c0 + tx]
            = (bf16_t)tile[tx][ty + i];
    }
}

// ---------------------------------------------------------------------------
// Prep: conv weights -> per-fragment swizzled bf16
// Fragment = 16(M) x 32(K); storage frag[lane*16 + e] with the ISA 16-bit A
// register layout (lanes 0-15: K{0..7,16..23}; lanes 16-31: K{8..15,24..31}).
// frag_id = (tap*16 + kb32)*32 + mt   (mt = out-channel/16)
// ---------------------------------------------------------------------------
__global__ void prep_wt_frag_kernel(const float* __restrict__ w, bf16_t* __restrict__ wtf) {
    int e = blockIdx.x * 256 + threadIdx.x;
    if (e >= 9 * 16 * 32 * 512) return;
    int frag = e >> 9, lane = (e >> 4) & 31, el = e & 15;
    int mt = frag & 31, kb32 = (frag >> 5) & 15, tap = frag >> 9;
    int o = mt * 16 + (lane & 15);
    int kk = (el < 8) ? ((lane >> 4) * 8 + el) : (16 + (lane >> 4) * 8 + (el - 8));
    int c = kb32 * 32 + kk;
    wtf[e] = (bf16_t)w[((size_t)o * CIN + c) * 9 + tap];
}

// Head weights (36 loc + 18 score rows, zero-pad to 64) -> fragment layout
__global__ void prep_whd_frag_kernel(const float* __restrict__ loc_w,
                                     const float* __restrict__ score_w,
                                     bf16_t* __restrict__ whf) {
    int e = blockIdx.x * 256 + threadIdx.x;
    if (e >= 16 * 4 * 512) return;
    int frag = e >> 9, lane = (e >> 4) & 31, el = e & 15;
    int mt = frag & 3, kb32 = frag >> 2;
    int o = mt * 16 + (lane & 15);
    int kk = (el < 8) ? ((lane >> 4) * 8 + el) : (16 + (lane >> 4) * 8 + (el - 8));
    int c = kb32 * 32 + kk;
    float v = 0.f;
    if (o < 36)      v = loc_w[(size_t)o * CIN + c];
    else if (o < 54) v = score_w[(size_t)(o - 36) * CIN + c];
    whf[e] = (bf16_t)v;
}

__global__ void anchors_kernel(float* __restrict__ anch) {
    int q = blockIdx.x * 256 + threadIdx.x;
    if (q >= NCAND) return;
    int a = q % NA, p = q / NA;
    int xg = p & 63, yg = p >> 6;
    const float ratios[3] = {0.5f, 1.0f, 2.0f};
    const float scales[3] = {8.0f, 16.0f, 32.0f};
    float r = ratios[a / 3], s = scales[a % 3];
    float hh = 16.0f * s * sqrtf(r);
    float ww = 16.0f * s * sqrtf(1.0f / r);
    float sx = xg * 16.0f, sy = yg * 16.0f;
    anch[q * 4 + 0] = sx - 0.5f * hh;
    anch[q * 4 + 1] = sy - 0.5f * ww;
    anch[q * 4 + 2] = sx + 0.5f * hh;
    anch[q * 4 + 3] = sy + 0.5f * ww;
}

__global__ void ridx_kernel(float* __restrict__ ridx) {
    int i = threadIdx.x;
    if (i < NB * N_POST) ridx[i] = (float)(i / N_POST);
}

// ---------------------------------------------------------------------------
// 3x3 conv as implicit GEMM, async-copy double-buffered LDS, bf16 WMMA.
// WG tile: 128 out-ch x one 64-pixel row. 8 waves = 4(M) x 2(N), wave 32x32.
// K = 9 taps x 512 ch, staged 64 at a time -> 72 stages, 8 WMMA / wave / stage.
// B tiles: GLOBAL_LOAD_ASYNC_TO_LDS_B128 from halo-padded NHWC bf16 (no VGPR
// round-trip, overlapped with WMMA). A fragments: coalesced 32B/lane global
// loads of pre-swizzled, L2-resident weights.
// ---------------------------------------------------------------------------
__global__ __launch_bounds__(256) void conv_wmma_kernel(
    const bf16_t* __restrict__ xhp, const bf16_t* __restrict__ wtf,
    const float* __restrict__ bias, bf16_t* __restrict__ hbuf) {

    __shared__ bf16_t sB[2][64 * 64];       // [buf][pixel][k], 8KB each

    const int tid  = threadIdx.x;
    const int mtb  = blockIdx.x * 8;        // m-tile base (16-wide tiles)
    const int y    = blockIdx.y;
    const int img  = blockIdx.z;
    const int wid  = tid >> 5, lane = tid & 31;
    const int wm   = wid & 3,  wn   = wid >> 2;
    const int half = lane >> 4, lm  = lane & 15;
    const int c8   = (tid & 7) * 8;         // B-load: k sub-offset
    const int px0  = tid >> 3;              // B-load: pixel 0..31

    const unsigned sb_base = (unsigned)(uintptr_t)&sB[0][0];

    v8f z = {0.f,0.f,0.f,0.f,0.f,0.f,0.f,0.f};
    v8f acc[2][2];
    #pragma unroll
    for (int mf = 0; mf < 2; ++mf)
        #pragma unroll
        for (int nf = 0; nf < 2; ++nf) acc[mf][nf] = z;

    auto issueB = [&](int s) {
        const int tap = s >> 3, kb = s & 7;
        const int dy = tap / 3 - 1, dxs = tap % 3 - 1;
        const int yy1 = y + dy + 1;                  // padded row, always valid
        const int b = s & 1;
        #pragma unroll
        for (int j = 0; j < 2; ++j) {
            const int px = px0 + j * 32;
            const int xx1 = px + dxs + 1;            // padded col, always valid
            const bf16_t* g =
                &xhp[(((size_t)img * PW + yy1) * PW + xx1) * CIN + kb * 64 + c8];
            const unsigned l = sb_base + (unsigned)((b * 4096 + px * 64 + c8) * 2);
            async_copy_b128(l, g);
        }
    };

    issueB(0);
    for (int s = 0; s < 72; ++s) {
        const int tap = s >> 3, kb = s & 7;
        wait_async0();                       // my stage-s copies landed
        __syncthreads();                     // everyone's landed; prev reads done
        if (s + 1 < 72) issueB(s + 1);       // fill other buffer during compute
        // A fragments: 32B contiguous per lane, coalesced, no LDS
        Frag16 fa[2][2];
        #pragma unroll
        for (int kh = 0; kh < 2; ++kh)
            #pragma unroll
            for (int mf = 0; mf < 2; ++mf) {
                const int frag = (tap * 16 + kb * 2 + kh) * 32 + mtb + wm * 2 + mf;
                fa[kh][mf].v = *(const v16bf*)(wtf + (size_t)frag * 512 + lane * 16);
            }
        const bf16_t* sb = sB[s & 1];
        #pragma unroll
        for (int kh = 0; kh < 2; ++kh) {
            #pragma unroll
            for (int nf = 0; nf < 2; ++nf) {
                Frag16 fb;
                const int n = wn * 32 + nf * 16 + lm;
                fb.h[0] = *(const v8bf*)(sb + n * 64 + kh * 32 + half * 8);
                fb.h[1] = *(const v8bf*)(sb + n * 64 + kh * 32 + 16 + half * 8);
                #pragma unroll
                for (int mf = 0; mf < 2; ++mf)
                    acc[mf][nf] = __builtin_amdgcn_wmma_f32_16x16x32_bf16(
                        false, fa[kh][mf].v, false, fb.v, (short)0, acc[mf][nf], false, false);
            }
        }
    }

    // epilogue: +bias, ReLU, packed v8bf stores, h in NHWC bf16
    #pragma unroll
    for (int mf = 0; mf < 2; ++mf) {
        const int cb = (mtb + wm * 2 + mf) * 16 + half * 8;
        float bv[8];
        #pragma unroll
        for (int v = 0; v < 8; ++v) bv[v] = bias[cb + v];
        #pragma unroll
        for (int nf = 0; nf < 2; ++nf) {
            const int p = y * 64 + wn * 32 + nf * 16 + lm;
            v8bf o8;
            #pragma unroll
            for (int v = 0; v < 8; ++v) {
                float val = acc[mf][nf][v] + bv[v];
                o8[v] = (bf16_t)(val > 0.f ? val : 0.f);
            }
            *(v8bf*)&hbuf[((size_t)img * NPIX + p) * CIN + cb] = o8;
        }
    }
}

// ---------------------------------------------------------------------------
// Head GEMM: [64 padded rows] x [32768 px], K=512, async-copy staging
// ---------------------------------------------------------------------------
__global__ __launch_bounds__(256) void head_wmma_kernel(
    const bf16_t* __restrict__ hbuf, const bf16_t* __restrict__ whf,
    const float* __restrict__ loc_b, const float* __restrict__ score_b,
    float* __restrict__ out_locs, float* __restrict__ out_scores,
    float* __restrict__ fg) {

    __shared__ bf16_t sB[2][64 * 64];

    const int tid  = threadIdx.x;
    const int pt   = blockIdx.x * 64;
    const int img  = pt >> 12;
    const int pbase= pt & (NPIX - 1);
    const int wid  = tid >> 5, lane = tid & 31;
    const int wm   = wid & 3,  wn   = wid >> 2;
    const int half = lane >> 4, lm  = lane & 15;
    const int c8   = (tid & 7) * 8;
    const int px0  = tid >> 3;

    const unsigned sb_base = (unsigned)(uintptr_t)&sB[0][0];

    v8f z = {0.f,0.f,0.f,0.f,0.f,0.f,0.f,0.f};
    v8f acc[2]; acc[0] = z; acc[1] = z;

    auto issueB = [&](int s) {
        const int b = s & 1;
        #pragma unroll
        for (int j = 0; j < 2; ++j) {
            const int px = px0 + j * 32;
            const bf16_t* g =
                &hbuf[((size_t)img * NPIX + pbase + px) * CIN + s * 64 + c8];
            const unsigned l = sb_base + (unsigned)((b * 4096 + px * 64 + c8) * 2);
            async_copy_b128(l, g);
        }
    };

    issueB(0);
    for (int s = 0; s < 8; ++s) {
        wait_async0();
        __syncthreads();
        if (s + 1 < 8) issueB(s + 1);
        Frag16 fa[2];
        #pragma unroll
        for (int kh = 0; kh < 2; ++kh) {
            const int frag = (s * 2 + kh) * 4 + wm;
            fa[kh].v = *(const v16bf*)(whf + (size_t)frag * 512 + lane * 16);
        }
        const bf16_t* sb = sB[s & 1];
        #pragma unroll
        for (int kh = 0; kh < 2; ++kh) {
            #pragma unroll
            for (int nf = 0; nf < 2; ++nf) {
                Frag16 fb;
                const int n = wn * 32 + nf * 16 + lm;
                fb.h[0] = *(const v8bf*)(sb + n * 64 + kh * 32 + half * 8);
                fb.h[1] = *(const v8bf*)(sb + n * 64 + kh * 32 + 16 + half * 8);
                acc[nf] = __builtin_amdgcn_wmma_f32_16x16x32_bf16(
                    false, fa[kh].v, false, fb.v, (short)0, acc[nf], false, false);
            }
        }
    }

    #pragma unroll
    for (int nf = 0; nf < 2; ++nf) {
        const int p = pbase + wn * 32 + nf * 16 + lm;
        #pragma unroll
        for (int v = 0; v < 8; ++v) {
            const int o = wm * 16 + half * 8 + v;
            float val = acc[nf][v];
            if (o < 36) {
                out_locs[((size_t)img * NCAND + (size_t)p * NA + (o >> 2)) * 4 + (o & 3)]
                    = val + loc_b[o];
            } else if (o < 54) {
                int oo = o - 36;
                float sc = val + score_b[oo];
                out_scores[((size_t)img * NCAND + (size_t)p * NA + (oo >> 1)) * 2 + (oo & 1)] = sc;
                if ((oo & 1) == 0) {   // fg = softmax[...,1] = sigmoid(s1 - s0)
                    float s1 = acc[nf][v + 1] + score_b[oo + 1];
                    fg[(size_t)img * NCAND + (size_t)p * NA + (oo >> 1)]
                        = 1.0f / (1.0f + __expf(sc - s1));
                }
            }
        }
    }
}

// ---------------------------------------------------------------------------
// Proposal stage (latency-trivial, f32 VALU)
// ---------------------------------------------------------------------------
__global__ void proposal_prep_kernel(const float* __restrict__ locs,
                                     const float* __restrict__ anch,
                                     const float* __restrict__ fg,
                                     const int* __restrict__ ihp,
                                     const int* __restrict__ iwp,
                                     float* __restrict__ roisall,
                                     float* __restrict__ smask) {
    int q = blockIdx.x * 256 + threadIdx.x;
    if (q >= NB * NCAND) return;
    int qi = q % NCAND;
    float a0 = anch[qi * 4 + 0], a1 = anch[qi * 4 + 1];
    float a2 = anch[qi * 4 + 2], a3 = anch[qi * 4 + 3];
    float w  = a2 - a0, h = a3 - a1;
    float cx = a0 + 0.5f * w, cy = a1 + 0.5f * h;
    const float* L = locs + (size_t)q * 4;
    float dx = L[0], dy = L[1], dw = L[2], dh = L[3];
    float ncx = dx * w + cx, ncy = dy * h + cy;
    float nw  = expf(dw) * w, nh = expf(dh) * h;
    float x1 = ncx - 0.5f * nw, y1 = ncy - 0.5f * nh;
    float x2 = ncx + 0.5f * nw, y2 = ncy + 0.5f * nh;
    float W = (float)iwp[0], H = (float)ihp[0];
    x1 = fminf(fmaxf(x1, 0.f), W); x2 = fminf(fmaxf(x2, 0.f), W);
    y1 = fminf(fmaxf(y1, 0.f), H); y2 = fminf(fmaxf(y2, 0.f), H);
    roisall[(size_t)q * 4 + 0] = x1; roisall[(size_t)q * 4 + 1] = y1;
    roisall[(size_t)q * 4 + 2] = x2; roisall[(size_t)q * 4 + 3] = y2;
    bool valid = ((x2 - x1) >= MIN_SZ) && ((y2 - y1) >= MIN_SZ);
    smask[q] = valid ? fg[q] : -INFINITY;
}

// 50 iterations of argmax (lowest-index tie-break, matching lax.top_k)
__global__ __launch_bounds__(256) void topk_kernel(float* __restrict__ s,
                                                   int* __restrict__ topi,
                                                   float* __restrict__ tops) {
    const int img = blockIdx.x;
    float* si_s = s + (size_t)img * NCAND;
    __shared__ float sv[256];
    __shared__ int   sx[256];
    for (int it = 0; it < N_PRE; ++it) {
        float best = -INFINITY; int bi = NCAND;
        for (int q = threadIdx.x; q < NCAND; q += 256) {
            float v = si_s[q];
            if (v > best || (v == best && q < bi)) { best = v; bi = q; }
        }
        sv[threadIdx.x] = best; sx[threadIdx.x] = bi;
        __syncthreads();
        for (int off = 128; off > 0; off >>= 1) {
            if (threadIdx.x < off) {
                float v = sv[threadIdx.x + off]; int qi = sx[threadIdx.x + off];
                if (v > sv[threadIdx.x] || (v == sv[threadIdx.x] && qi < sx[threadIdx.x])) {
                    sv[threadIdx.x] = v; sx[threadIdx.x] = qi;
                }
            }
            __syncthreads();
        }
        if (threadIdx.x == 0) {
            topi[img * N_PRE + it] = sx[0];
            tops[img * N_PRE + it] = sv[0];
            si_s[sx[0]] = -INFINITY;
        }
        __syncthreads();
    }
}

__global__ void nms_kernel(const float* __restrict__ roisall,
                           const int* __restrict__ topi,
                           const float* __restrict__ tops,
                           float* __restrict__ out_rois) {
    if (threadIdx.x != 0) return;
    const int img = blockIdx.x;
    float bx[N_PRE][4], sc[N_PRE], area[N_PRE];
    bool keep[N_PRE];
    for (int i = 0; i < N_PRE; ++i) {
        int q = topi[img * N_PRE + i];
        for (int t = 0; t < 4; ++t)
            bx[i][t] = roisall[((size_t)img * NCAND + q) * 4 + t];
        sc[i] = tops[img * N_PRE + i];
        keep[i] = (sc[i] > -1e38f);
        area[i] = (bx[i][2] - bx[i][0]) * (bx[i][3] - bx[i][1]);
    }
    for (int i = 0; i < N_PRE; ++i) {
        if (!keep[i]) continue;
        for (int j = i + 1; j < N_PRE; ++j) {
            float xx1 = fmaxf(bx[i][0], bx[j][0]);
            float yy1 = fmaxf(bx[i][1], bx[j][1]);
            float xx2 = fminf(bx[i][2], bx[j][2]);
            float yy2 = fminf(bx[i][3], bx[j][3]);
            float inter = fmaxf(xx2 - xx1, 0.f) * fmaxf(yy2 - yy1, 0.f);
            float iou = inter / (area[i] + area[j] - inter + 1e-9f);
            if (iou > NMS_IOU) keep[j] = false;
        }
    }
    int order[N_PRE]; int nk = 0;
    for (int i = 0; i < N_PRE; ++i) if (keep[i])  order[nk++] = i;
    int tot = nk;
    for (int i = 0; i < N_PRE; ++i) if (!keep[i]) order[tot++] = i;
    int nkc = nk > 0 ? nk : 1;
    for (int pos = 0; pos < N_POST; ++pos) {
        int sel = order[pos < nkc ? pos : pos % nkc];
        for (int t = 0; t < 4; ++t)
            out_rois[((size_t)img * N_POST + pos) * 4 + t] = bx[sel][t];
    }
}

// ---------------------------------------------------------------------------
// Host-side launcher
// ---------------------------------------------------------------------------
extern "C" void kernel_launch(void* const* d_in, const int* in_sizes, int n_in,
                              void* d_out, int out_size, void* d_ws, size_t ws_size,
                              hipStream_t stream) {
    const float* x        = (const float*)d_in[0];
    const float* conv1_w  = (const float*)d_in[1];
    const float* conv1_b  = (const float*)d_in[2];
    const float* score_w  = (const float*)d_in[3];
    const float* score_b  = (const float*)d_in[4];
    const float* loc_w    = (const float*)d_in[5];
    const float* loc_b    = (const float*)d_in[6];
    const int*   img_h    = (const int*)d_in[7];
    const int*   img_w    = (const int*)d_in[8];
    float* out = (float*)d_out;

    // workspace carving (byte offsets)
    char* w = (char*)d_ws;
    bf16_t* wtf     = (bf16_t*)(w + 0);            // 4,718,592 B
    bf16_t* whf     = (bf16_t*)(w + 4718592);      //    65,536 B
    bf16_t* xhp     = (bf16_t*)(w + 4784128);      // padded NHWC x bf16 = 35,684,352 B
    bf16_t* hbuf    = (bf16_t*)(w + 40468480);     // NHWC h bf16        = 33,554,432 B
    float*  fg      = (float*) (w + 74022912);     // 1,179,648 B
    float*  smask   = (float*) (w + 75202560);     // 1,179,648 B
    float*  roisall = (float*) (w + 76382208);     // 4,718,592 B
    int*    topi    = (int*)   (w + 81100800);     // 8*50 int
    float*  tops    = (float*) (w + 81102848);     // 8*50 f32  (end ~81.1 MB)

    // 1) prep
    const int xhp_n16 = (NB * PW * PW * CIN * 2) / 16;   // 16B chunks
    zero_xhp_kernel<<<(xhp_n16 + 255) / 256, 256, 0, stream>>>((v4u*)xhp, xhp_n16);
    prep_xh_kernel<<<dim3(NPIX / 32, CIN / 32, NB), 256, 0, stream>>>(x, xhp);
    prep_wt_frag_kernel<<<(9 * 16 * 32 * 512 + 255) / 256, 256, 0, stream>>>(conv1_w, wtf);
    prep_whd_frag_kernel<<<(16 * 4 * 512 + 255) / 256, 256, 0, stream>>>(loc_w, score_w, whf);
    anchors_kernel<<<(NCAND + 255) / 256, 256, 0, stream>>>(out + OFF_ANCH);
    ridx_kernel<<<1, 128, 0, stream>>>(out + OFF_RIDX);

    // 2) conv (implicit GEMM, bf16 WMMA, async-copy double-buffered LDS)
    conv_wmma_kernel<<<dim3(COUT / 128, FH, NB), 256, 0, stream>>>(xhp, wtf, conv1_b, hbuf);

    // 3) head GEMM -> locs/scores/fg
    head_wmma_kernel<<<dim3((NB * NPIX) / 64), 256, 0, stream>>>(
        hbuf, whf, loc_b, score_b, out + OFF_LOCS, out + OFF_SCORES, fg);

    // 4) proposals
    proposal_prep_kernel<<<(NB * NCAND + 255) / 256, 256, 0, stream>>>(
        out + OFF_LOCS, out + OFF_ANCH, fg, img_h, img_w, roisall, smask);
    topk_kernel<<<NB, 256, 0, stream>>>(smask, topi, tops);
    nms_kernel<<<NB, 32, 0, stream>>>(roisall, topi, tops, out + OFF_ROIS);
}
